// MessagePassingLayer_65420941853023
// MI455X (gfx1250) — matrix operand order repaired and checked by reading, not compiled, not run on hardware
//
#include <hip/hip_runtime.h>
#include <hip/hip_bf16.h>

// Problem constants (match reference)
#define B_  8
#define A_  256
#define N_  64
#define F_  128
#define NB_ 20

typedef __bf16 bf16_t;
typedef __attribute__((ext_vector_type(16))) __bf16 v16bf;
typedef __attribute__((ext_vector_type(8)))  __bf16 v8bf;
typedef __attribute__((ext_vector_type(8)))  float  v8f;

// LDS strides (halves); multiples of 8 -> 16B aligned chunks
#define SW 136
#define SA 136

// silu via hardware reciprocal (v_rcp_f32) -- avoids the IEEE div expansion
__device__ __forceinline__ float silu_f(float x) {
  return x * __builtin_amdgcn_rcpf(1.0f + __expf(-x));
}

// cutoff: P=9 -> 1 - 55 r^9 + 99 r^10 - 45 r^11, zero for d >= 5
__device__ __forceinline__ float poly_cutoff(float d) {
  float r  = d * 0.2f;
  float r2 = r * r, r4 = r2 * r2, r8 = r4 * r4;
  float r9 = r8 * r, r10 = r9 * r, r11 = r10 * r;
  float c  = 1.0f - 55.0f * r9 + 99.0f * r10 - 45.0f * r11;
  return (d < 5.0f) ? c : 0.0f;
}

// ---- WMMA fragment loaders (CDNA5 16-bit layouts, ISA 7.12.2) ----
// A 16x32 (MxK): lanes 0-15 -> K {0..7, 16..23}; lanes 16-31 -> K {8..15, 24..31}
// Rows are contiguous in LDS -> two 16B ds_load_b128 per fragment.
__device__ __forceinline__ v16bf load_a_frag(const bf16_t* A, int mt, int kt, int lane) {
  const bf16_t* p = A + (mt * 16 + (lane & 15)) * SA + kt * 32 + ((lane < 16) ? 0 : 8);
  union { v16bf v; v8bf h[2]; } u;
  u.h[0] = *(const v8bf*)(p);
  u.h[1] = *(const v8bf*)(p + 16);
  return u.v;
}
// B 32x16 (KxN): lanes 0-15 -> K {0..7, 16..23}; lanes 16-31 -> K {8..15, 24..31}; N = lane&15.
// Weight staged TRANSPOSED in LDS (wT[n][k]) so K is contiguous -> two b128 loads.
__device__ __forceinline__ v16bf load_b_frag(const bf16_t* WT, int kt, int nt, int lane) {
  const bf16_t* p = WT + (nt * 16 + (lane & 15)) * SW + kt * 32 + ((lane < 16) ? 0 : 8);
  union { v16bf v; v8bf h[2]; } u;
  u.h[0] = *(const v8bf*)(p);
  u.h[1] = *(const v8bf*)(p + 16);
  return u.v;
}

// C = A(64x128 bf16, LDS) @ W(128x128, LDS transposed), f32 accum.
// 8 waves: wave w owns N-tile w; loops 4 M-tiles; K = 4 x 32.
// epi(row0, col, acc): acc[i] is element (row0+i, col).
template <typename EPI>
__device__ __forceinline__ void gemm_64x128(const bf16_t* A, const bf16_t* WT, EPI epi) {
  const int lane = threadIdx.x & 31;
  const int nt   = threadIdx.x >> 5;   // 0..7
#pragma unroll
  for (int mt = 0; mt < 4; ++mt) {
    v8f acc = {};
#pragma unroll
    for (int kt = 0; kt < 4; ++kt) {
      v16bf a = load_a_frag(A, mt, kt, lane);
      v16bf b = load_b_frag(WT, kt, nt, lane);
      acc = __builtin_amdgcn_wmma_f32_16x16x32_bf16(false, a, false, b, (short)0, acc,
                                                    false, false);
    }
    epi(mt * 16 + ((lane < 16) ? 0 : 8), nt * 16 + (lane & 15), acc);
  }
}

// stage 128x128 f32 weight from global -> bf16 LDS, TRANSPOSED: wT[c*SW + r] = W[r][c]
__device__ __forceinline__ void stage_wT(const float* Wg, bf16_t* wT) {
  for (int idx = threadIdx.x; idx < F_ * F_; idx += 256) {
    int r = idx >> 7, c = idx & 127;
    wT[c * SW + r] = (bf16_t)Wg[idx];
  }
}

// =========================== Kernel 1: imn = MLP(inv_node) ===========================
__global__ void k1_node_mlp(const float* __restrict__ inv_node,
                            const float* __restrict__ Wn1, const float* __restrict__ bn1,
                            const float* __restrict__ Wn2, const float* __restrict__ bn2,
                            float* __restrict__ ws_imn) {
  extern __shared__ char smem[];
  bf16_t* w   = (bf16_t*)smem;
  bf16_t* act = w + F_ * SW;
  bf16_t* hid = act + 64 * SA;
  const int node0 = blockIdx.x * 64;

  for (int idx = threadIdx.x; idx < 64 * F_; idx += 256) {
    int r = idx >> 7, c = idx & 127;
    act[r * SA + c] = (bf16_t)inv_node[(node0 + r) * F_ + c];
  }
  stage_wT(Wn1, w);
  __syncthreads();
  gemm_64x128(act, w, [&](int row0, int col, v8f acc) {
    float bias = bn1[col];
#pragma unroll
    for (int i = 0; i < 8; ++i) {
      float h = acc[i] + bias;
      hid[(row0 + i) * SA + col] = (bf16_t)silu_f(h);
    }
  });
  __syncthreads();
  stage_wT(Wn2, w);
  __syncthreads();
  gemm_64x128(hid, w, [&](int row0, int col, v8f acc) {
    float bias = bn2[col];
#pragma unroll
    for (int i = 0; i < 8; ++i)
      ws_imn[(node0 + row0 + i) * F_ + col] = acc[i] + bias;
  });
}

// =========================== Kernel 2: per-atom edge work ===========================
__global__ void k2_edge(const float* __restrict__ inv_node, const float* __restrict__ inv_edge,
                        const float* __restrict__ dist, const float* __restrict__ dvec,
                        const int* __restrict__ nbr_g, const float* __restrict__ mask_g,
                        const float* __restrict__ eqF_in, const float* __restrict__ eqf_in,
                        const float* __restrict__ eqdr_in,
                        const float* __restrict__ Wme, const float* __restrict__ bme,
                        const float* __restrict__ Wc,
                        const float* __restrict__ Wf1, const float* __restrict__ bf1v,
                        const float* __restrict__ Wf2, const float* __restrict__ bf2v,
                        const float* __restrict__ We1, const float* __restrict__ We2,
                        const float* __restrict__ ws_imn, float* __restrict__ ws_eqf,
                        float* __restrict__ out_inv, float* __restrict__ out_eqF,
                        float* __restrict__ out_eqf, float* __restrict__ out_eqdr) {
  extern __shared__ char smem[];
  float* edge_s   = (float*)smem;            // 64*20
  float* invupd   = edge_s + N_ * NB_;       // 128   (contiguous zero region start)
  float* emFdot   = invupd + F_;             // 64
  float* emFm     = emFdot + N_;             // 64*3
  float* eqf_acc  = emFm + N_ * 3;           // 3*128
  float* eqdr_acc = eqf_acc + 3 * F_;        // 3*128
  float* imns     = eqdr_acc + 3 * F_;       // 128
  float* cut_s    = imns + F_;               // 64
  float* mask_s   = cut_s + N_;              // 64
  int*   nbr_s    = (int*)(mask_s + N_);     // 64
  bf16_t* w   = (bf16_t*)(nbr_s + N_);
  bf16_t* act = w + F_ * SW;
  bf16_t* hid = act + 64 * SA;

  const int tid = threadIdx.x;
  const int ba  = blockIdx.x;        // b*A + a
  const int b   = ba >> 8;           // A_ == 256

  // zero f32 accumulators (invupd..eqdr_acc contiguous: 128+64+192+384+384 = 1152)
  for (int i = tid; i < 1152; i += 256) invupd[i] = 0.0f;
  if (tid < N_) {
    nbr_s[tid]  = nbr_g[ba * N_ + tid];
    mask_s[tid] = mask_g[ba * N_ + tid];
    cut_s[tid]  = poly_cutoff(dist[ba * N_ + tid]);
  }
  for (int i = tid; i < N_ * NB_; i += 256) edge_s[i] = inv_edge[ba * N_ * NB_ + i];
  if (tid < F_) imns[tid] = ws_imn[ba * F_ + tid];
  __syncthreads();

  // ---- imsg = (edge@Wme + bme)*cutoff * imn[nbr] * imn[self]  -> act (bf16) ----
  {
    const int f    = tid & 127;
    const int half = tid >> 7;
    float wme[NB_];
#pragma unroll
    for (int k = 0; k < NB_; ++k) wme[k] = Wme[k * F_ + f];
    const float bmev = bme[f];
    const float self = imns[f];
    float invsum = 0.0f;
    for (int n = half * 32; n < half * 32 + 32; ++n) {
      float s = bmev;
#pragma unroll
      for (int k = 0; k < NB_; ++k) s += edge_s[n * NB_ + k] * wme[k];
      s *= cut_s[n];
      float g   = ws_imn[(b * A_ + nbr_s[n]) * F_ + f];
      float msg = s * g * self;
      act[n * SA + f] = (bf16_t)msg;
      invsum += msg * mask_s[n];
    }
    atomicAdd(&invupd[f], invsum);
  }
  __syncthreads();

  // ---- emFdot[n] = imsg[n,:] . Wc ----
  {
    const int row = tid >> 2, part = tid & 3;
    float s = 0.0f;
    for (int c = part * 32; c < part * 32 + 32; ++c)
      s += (float)act[row * SA + c] * Wc[c];
    atomicAdd(&emFdot[row], s);
  }
  __syncthreads();

  // ---- emFm[n][d] = emFdot[n]*dvec[n][d]*mask[n] ----
  if (tid < 3 * N_) {
    int n = tid & 63, d = tid >> 6;
    emFm[n * 3 + d] = emFdot[n] * dvec[(ba * N_ + n) * 3 + d] * mask_s[n];
  }
  __syncthreads();

  if (tid < 3) {
    float s = eqF_in[ba * 3 + tid];
    for (int n = 0; n < N_; ++n) s += emFm[n * 3 + tid];
    out_eqF[ba * 3 + tid] = s;
  }
  if (tid < F_) out_inv[ba * F_ + tid] = inv_node[ba * F_ + tid] + invupd[tid];

  // ---- Wf MLP + eqf_upd reduction ----
  stage_wT(Wf1, w);
  __syncthreads();
  gemm_64x128(act, w, [&](int row0, int col, v8f acc) {
    float bias = bf1v[col];
#pragma unroll
    for (int i = 0; i < 8; ++i)
      hid[(row0 + i) * SA + col] = (bf16_t)silu_f(acc[i] + bias);
  });
  __syncthreads();
  stage_wT(Wf2, w);
  __syncthreads();
  gemm_64x128(hid, w, [&](int row0, int col, v8f acc) {
    float bias = bf2v[col];
#pragma unroll
    for (int i = 0; i < 8; ++i) {
      float v = acc[i] + bias;
      int row = row0 + i;
#pragma unroll
      for (int d = 0; d < 3; ++d)
        atomicAdd(&eqf_acc[d * F_ + col], v * emFm[row * 3 + d]);
    }
  });
  __syncthreads();
  for (int i = tid; i < 3 * F_; i += 256) {
    float v = eqf_acc[i];
    ws_eqf[ba * 3 * F_ + i] = v;
    out_eqf[ba * 3 * F_ + i] = eqf_in[ba * 3 * F_ + i] + v;
  }

  // ---- We MLP + eq_dr-gather reduction ----
  stage_wT(We1, w);
  __syncthreads();
  gemm_64x128(act, w, [&](int row0, int col, v8f acc) {
#pragma unroll
    for (int i = 0; i < 8; ++i)
      hid[(row0 + i) * SA + col] = (bf16_t)silu_f(acc[i]);
  });
  __syncthreads();
  stage_wT(We2, w);
  __syncthreads();
  gemm_64x128(hid, w, [&](int row0, int col, v8f acc) {
#pragma unroll
    for (int i = 0; i < 8; ++i) {
      int row = row0 + i;
      float v = acc[i] * mask_s[row];
      int gnode = b * A_ + nbr_s[row];
#pragma unroll
      for (int d = 0; d < 3; ++d) {
        float g = eqdr_in[(gnode * 3 + d) * F_ + col];
        atomicAdd(&eqdr_acc[d * F_ + col], v * g);
      }
    }
  });
  __syncthreads();
  for (int i = tid; i < 3 * F_; i += 256)
    out_eqdr[ba * 3 * F_ + i] = eqdr_in[ba * 3 * F_ + i] + eqdr_acc[i];
}

// =========================== Kernel 3: Ws / Wi finalize ===========================
__global__ void k3_final(const float* __restrict__ Ws1, const float* __restrict__ bs1,
                         const float* __restrict__ Ws2, const float* __restrict__ bs2,
                         const float* __restrict__ Wi1, const float* __restrict__ bi1,
                         const float* __restrict__ Wi2, const float* __restrict__ bi2,
                         const float* __restrict__ ws_eqf,
                         float* __restrict__ out_inv, const float* __restrict__ out_eqf,
                         float* __restrict__ out_eqdr) {
  extern __shared__ char smem[];
  bf16_t* w   = (bf16_t*)smem;
  bf16_t* act = w + F_ * SW;
  bf16_t* hid = act + 64 * SA;
  const int node0 = blockIdx.x * 64;

  for (int idx = threadIdx.x; idx < 64 * F_; idx += 256) {
    int r = idx >> 7, c = idx & 127;
    act[r * SA + c] = (bf16_t)out_inv[(node0 + r) * F_ + c];  // inv_node_new
  }
  // ---- eq_dr += silu-MLP_s(inv_new) * eqf_upd ----
  stage_wT(Ws1, w);
  __syncthreads();
  gemm_64x128(act, w, [&](int row0, int col, v8f acc) {
    float bias = bs1[col];
#pragma unroll
    for (int i = 0; i < 8; ++i)
      hid[(row0 + i) * SA + col] = (bf16_t)silu_f(acc[i] + bias);
  });
  __syncthreads();
  stage_wT(Ws2, w);
  __syncthreads();
  gemm_64x128(hid, w, [&](int row0, int col, v8f acc) {
    float bias = bs2[col];
#pragma unroll
    for (int i = 0; i < 8; ++i) {
      float s = acc[i] + bias;
      int node = node0 + row0 + i;
#pragma unroll
      for (int d = 0; d < 3; ++d) {
        size_t idx = ((size_t)node * 3 + d) * F_ + col;
        out_eqdr[idx] += s * ws_eqf[idx];
      }
    }
  });
  __threadfence();
  __syncthreads();
  // ---- inv_final = inv_new - MLP_i(inv_new) * sum_d(eq_f * eq_dr) ----
  stage_wT(Wi1, w);
  __syncthreads();
  gemm_64x128(act, w, [&](int row0, int col, v8f acc) {
    float bias = bi1[col];
#pragma unroll
    for (int i = 0; i < 8; ++i)
      hid[(row0 + i) * SA + col] = (bf16_t)silu_f(acc[i] + bias);
  });
  __syncthreads();
  stage_wT(Wi2, w);
  __syncthreads();
  gemm_64x128(hid, w, [&](int row0, int col, v8f acc) {
    float bias = bi2[col];
#pragma unroll
    for (int i = 0; i < 8; ++i) {
      float t = acc[i] + bias;
      int node = node0 + row0 + i;
      float dot = 0.0f;
#pragma unroll
      for (int d = 0; d < 3; ++d) {
        size_t idx = ((size_t)node * 3 + d) * F_ + col;
        dot += out_eqf[idx] * out_eqdr[idx];
      }
      size_t oidx = (size_t)node * F_ + col;
      out_inv[oidx] = out_inv[oidx] - t * dot;
    }
  });
}

extern "C" void kernel_launch(void* const* d_in, const int* in_sizes, int n_in,
                              void* d_out, int out_size, void* d_ws, size_t ws_size,
                              hipStream_t stream) {
  (void)in_sizes; (void)n_in; (void)out_size; (void)ws_size;
  const float* inv_node = (const float*)d_in[0];
  const float* inv_edge = (const float*)d_in[1];
  const float* dist     = (const float*)d_in[2];
  const float* dvec     = (const float*)d_in[3];
  const int*   nbr      = (const int*)d_in[4];
  const float* mask     = (const float*)d_in[5];
  const float* eqF      = (const float*)d_in[6];
  const float* eqf      = (const float*)d_in[7];
  const float* eqdr     = (const float*)d_in[8];
  const float* Wme = (const float*)d_in[9];
  const float* bme = (const float*)d_in[10];
  const float* Wn1 = (const float*)d_in[11];
  const float* bn1 = (const float*)d_in[12];
  const float* Wn2 = (const float*)d_in[13];
  const float* bn2 = (const float*)d_in[14];
  const float* Wc  = (const float*)d_in[15];
  const float* Wf1 = (const float*)d_in[16];
  const float* bf1 = (const float*)d_in[17];
  const float* Wf2 = (const float*)d_in[18];
  const float* bf2 = (const float*)d_in[19];
  const float* Ws1 = (const float*)d_in[20];
  const float* bs1 = (const float*)d_in[21];
  const float* Ws2 = (const float*)d_in[22];
  const float* bs2 = (const float*)d_in[23];
  const float* We1 = (const float*)d_in[24];
  const float* We2 = (const float*)d_in[25];
  const float* Wi1 = (const float*)d_in[26];
  const float* bi1 = (const float*)d_in[27];
  const float* Wi2 = (const float*)d_in[28];
  const float* bi2 = (const float*)d_in[29];

  float* out_inv  = (float*)d_out;
  float* out_eqF  = out_inv + B_ * A_ * F_;
  float* out_eqf  = out_eqF + B_ * A_ * 3;
  float* out_eqdr = out_eqf + B_ * A_ * 3 * F_;

  float* ws_imn = (float*)d_ws;                 // (B*A, F)
  float* ws_eqf = ws_imn + B_ * A_ * F_;        // (B*A, 3, F)

  const size_t lds_gemm = (size_t)(F_ * SW + 2 * 64 * SA) * sizeof(bf16_t);      // ~68 KB
  const size_t lds_k2   = 2688 * sizeof(float) + 64 * sizeof(int) + lds_gemm;    // ~79 KB

  (void)hipFuncSetAttribute((const void*)k1_node_mlp,
                            hipFuncAttributeMaxDynamicSharedMemorySize, (int)lds_gemm);
  (void)hipFuncSetAttribute((const void*)k2_edge,
                            hipFuncAttributeMaxDynamicSharedMemorySize, (int)lds_k2);
  (void)hipFuncSetAttribute((const void*)k3_final,
                            hipFuncAttributeMaxDynamicSharedMemorySize, (int)lds_gemm);

  k1_node_mlp<<<(B_ * A_) / 64, 256, lds_gemm, stream>>>(inv_node, Wn1, bn1, Wn2, bn2, ws_imn);

  k2_edge<<<B_ * A_, 256, lds_k2, stream>>>(inv_node, inv_edge, dist, dvec, nbr, mask,
                                            eqF, eqf, eqdr, Wme, bme, Wc,
                                            Wf1, bf1, Wf2, bf2, We1, We2,
                                            ws_imn, ws_eqf,
                                            out_inv, out_eqF, out_eqf, out_eqdr);

  k3_final<<<(B_ * A_) / 64, 256, lds_gemm, stream>>>(Ws1, bs1, Ws2, bs2, Wi1, bi1, Wi2, bi2,
                                                      ws_eqf, out_inv, out_eqf, out_eqdr);
}
